// RGCN_41377714929863
// MI455X (gfx1250) — compile-verified, matrix operand order checked by reading, not run on hardware
//
#include <hip/hip_runtime.h>
#include <hip/hip_bf16.h>

typedef __attribute__((ext_vector_type(16))) __bf16 v16bf;
typedef __attribute__((ext_vector_type(8)))  __bf16 v8bf;
typedef __attribute__((ext_vector_type(8)))  float  v8f;

#define H_DIM   128
#define OUT_DIM 16
#define N_BASES 4
#define NC1 (N_BASES * H_DIM + H_DIM)     // 640: [basis-cat | loop]
#define NC2 (N_BASES * OUT_DIM + OUT_DIM) // 80

__device__ __forceinline__ __bf16 f2bf(float f) {
  union { float f; unsigned u; } c; c.f = f;
  unsigned u = c.u + 0x7FFFu + ((c.u >> 16) & 1u);   // round-to-nearest-even
  union { unsigned short s; __bf16 b; } r; r.s = (unsigned short)(u >> 16);
  return r.b;
}

// ---- gather + convert: Abf[n][i] = bf16(emb[node_ids[n]][i])
__global__ void gather_cvt_bf16(const float* __restrict__ emb,
                                const int* __restrict__ ids,
                                __bf16* __restrict__ abf, int nNodes) {
  int i = blockIdx.x * blockDim.x + threadIdx.x;
  if (i >= nNodes * H_DIM) return;
  int node = i >> 7, ch = i & 127;
  abf[i] = f2bf(emb[(size_t)ids[node] * H_DIM + ch]);
}

// ---- pack basis1 [4,128,128] + loop_w1 [128,128] -> BT1 [640][128] bf16 (K contiguous)
__global__ void pack_bt1(const float* __restrict__ basis1,
                         const float* __restrict__ loop_w1,
                         __bf16* __restrict__ bt) {
  int idx = blockIdx.x * blockDim.x + threadIdx.x;
  if (idx >= NC1 * H_DIM) return;
  int c = idx / H_DIM, i = idx % H_DIM;
  float v;
  if (c < N_BASES * H_DIM) {
    int b = c >> 7, o = c & 127;
    v = basis1[((size_t)b * H_DIM + i) * H_DIM + o];
  } else {
    v = loop_w1[(size_t)i * H_DIM + (c - N_BASES * H_DIM)];
  }
  bt[(size_t)c * H_DIM + i] = f2bf(v);
}

// ---- pack basis2 [4,128,16] + loop_w2 [128,16] -> BT2 [80][128] bf16
__global__ void pack_bt2(const float* __restrict__ basis2,
                         const float* __restrict__ loop_w2,
                         __bf16* __restrict__ bt) {
  int idx = blockIdx.x * blockDim.x + threadIdx.x;
  if (idx >= NC2 * H_DIM) return;
  int c = idx / H_DIM, i = idx % H_DIM;
  float v;
  if (c < N_BASES * OUT_DIM) {
    int b = c >> 4, o = c & 15;
    v = basis2[((size_t)b * H_DIM + i) * OUT_DIM + o];
  } else {
    v = loop_w2[(size_t)i * OUT_DIM + (c - N_BASES * OUT_DIM)];
  }
  bt[(size_t)c * H_DIM + i] = f2bf(v);
}

// ---- WMMA GEMM: OUT[M x Ncols](f32) = A[M x 128](bf16) * BT[Ncols x 128](bf16)^T
// one wave per 16x16 tile; K=128 fully unrolled as 4x v_wmma_f32_16x16x32_bf16
__global__ __launch_bounds__(32)
void wmma_gemm_k128(const __bf16* __restrict__ A, const __bf16* __restrict__ BT,
                    float* __restrict__ OUT, int M, int Ncols) {
  const int lane = threadIdx.x;    // 0..31
  const int half = lane >> 4;      // 0/1
  const int l    = lane & 15;
  const int m0 = blockIdx.x * 16;
  const int n0 = blockIdx.y * 16;

  int arow = m0 + l; if (arow >= M) arow = M - 1;
  const __bf16* Ap = A  + (size_t)arow * H_DIM + half * 8;
  const __bf16* Bp = BT + (size_t)(n0 + l) * H_DIM + half * 8;

  v8f acc = {};
#pragma unroll
  for (int k0 = 0; k0 < H_DIM; k0 += 32) {
    // 16-bit WMMA lane layout: K = k0 + 16*(e>>3) + 8*half + (e&7)
    v8bf alo = *(const v8bf*)(Ap + k0);
    v8bf ahi = *(const v8bf*)(Ap + k0 + 16);
    v8bf blo = *(const v8bf*)(Bp + k0);
    v8bf bhi = *(const v8bf*)(Bp + k0 + 16);
    v16bf a = __builtin_shufflevector(alo, ahi, 0, 1, 2, 3, 4, 5, 6, 7,
                                      8, 9, 10, 11, 12, 13, 14, 15);
    v16bf b = __builtin_shufflevector(blo, bhi, 0, 1, 2, 3, 4, 5, 6, 7,
                                      8, 9, 10, 11, 12, 13, 14, 15);
    acc = __builtin_amdgcn_wmma_f32_16x16x32_bf16(
        false, a, false, b, (short)0, acc, false, false);
  }

  // C/D layout: VGPR g -> row m0 + g + 8*half, col n0 + l
  if (m0 + 16 <= M) {
    float* p = OUT + (size_t)(m0 + 8 * half) * Ncols + n0 + l;
#pragma unroll
    for (int g = 0; g < 8; ++g) p[(size_t)g * Ncols] = acc[g];
  } else {
#pragma unroll
    for (int g = 0; g < 8; ++g) {
      int row = m0 + g + 8 * half;
      if (row < M) OUT[(size_t)row * Ncols + n0 + l] = acc[g];
    }
  }
}

// ---- layer-1 edge scatter: wave per edge, lane owns 4 channels
__global__ void edge_scatter1(const int* __restrict__ src, const int* __restrict__ dst,
                              const int* __restrict__ etype,
                              const float* __restrict__ xb,   // [N,640]
                              const float* __restrict__ comp, // [90,4]
                              float* __restrict__ agg,        // [N,128]
                              float* __restrict__ deg, int nEdges) {
  int lane = threadIdx.x & 31;
  int edge = (blockIdx.x * blockDim.x + threadIdx.x) >> 5;
  if (edge >= nEdges) return;
  int s = src[edge], d = dst[edge], r = etype[edge];
  const float* w = comp + (size_t)r * N_BASES;
  float w0 = w[0], w1 = w[1], w2 = w[2], w3 = w[3];
  const float* x = xb + (size_t)s * NC1 + lane * 4;
  float4 v0 = *(const float4*)(x + 0 * H_DIM);
  float4 v1 = *(const float4*)(x + 1 * H_DIM);
  float4 v2 = *(const float4*)(x + 2 * H_DIM);
  float4 v3 = *(const float4*)(x + 3 * H_DIM);
  float4 m;
  m.x = w0 * v0.x + w1 * v1.x + w2 * v2.x + w3 * v3.x;
  m.y = w0 * v0.y + w1 * v1.y + w2 * v2.y + w3 * v3.y;
  m.z = w0 * v0.z + w1 * v1.z + w2 * v2.z + w3 * v3.z;
  m.w = w0 * v0.w + w1 * v1.w + w2 * v2.w + w3 * v3.w;
  float* o = agg + (size_t)d * H_DIM + lane * 4;
  atomicAdd(o + 0, m.x); atomicAdd(o + 1, m.y);
  atomicAdd(o + 2, m.z); atomicAdd(o + 3, m.w);
  if (lane == 0) atomicAdd(deg + d, 1.0f);
}

// ---- h = relu(agg/max(deg,1) + loop_cols + bias1), written as bf16 (feeds GEMM-2)
__global__ void finish1(const float* __restrict__ agg, const float* __restrict__ deg,
                        const float* __restrict__ xb, const float* __restrict__ bias1,
                        __bf16* __restrict__ hbf, int nNodes) {
  int i = blockIdx.x * blockDim.x + threadIdx.x;
  if (i >= nNodes * H_DIM) return;
  int node = i >> 7, ch = i & 127;
  float dg = deg[node]; dg = dg > 1.f ? dg : 1.f;
  float v = agg[i] / dg + xb[(size_t)node * NC1 + N_BASES * H_DIM + ch] + bias1[ch];
  hbf[i] = f2bf(v > 0.f ? v : 0.f);
}

// ---- layer-2 edge scatter: thread per (edge, out-channel)
__global__ void edge_scatter2(const int* __restrict__ src, const int* __restrict__ dst,
                              const int* __restrict__ etype,
                              const float* __restrict__ xb,   // [N,80]
                              const float* __restrict__ comp, // [90,4]
                              float* __restrict__ agg,        // [N,16]
                              int nEdges) {
  int idx = blockIdx.x * blockDim.x + threadIdx.x;
  if (idx >= nEdges * OUT_DIM) return;
  int e = idx >> 4, o = idx & 15;
  int s = src[e], d = dst[e], r = etype[e];
  const float* w = comp + (size_t)r * N_BASES;
  const float* x = xb + (size_t)s * NC2;
  float v = w[0] * x[o] + w[1] * x[OUT_DIM + o] +
            w[2] * x[2 * OUT_DIM + o] + w[3] * x[3 * OUT_DIM + o];
  atomicAdd(agg + (size_t)d * OUT_DIM + o, v);
}

// ---- out = agg/max(deg,1) + loop_cols + bias2
__global__ void finish2(const float* __restrict__ agg, const float* __restrict__ deg,
                        const float* __restrict__ xb, const float* __restrict__ bias2,
                        float* __restrict__ out, int nNodes) {
  int i = blockIdx.x * blockDim.x + threadIdx.x;
  if (i >= nNodes * OUT_DIM) return;
  int node = i >> 4, o = i & 15;
  float dg = deg[node]; dg = dg > 1.f ? dg : 1.f;
  out[i] = agg[i] / dg + xb[(size_t)node * NC2 + N_BASES * OUT_DIM + o] + bias2[o];
}

extern "C" void kernel_launch(void* const* d_in, const int* in_sizes, int n_in,
                              void* d_out, int out_size, void* d_ws, size_t ws_size,
                              hipStream_t stream) {
  const int*   node_ids = (const int*)  d_in[0];
  const int*   src      = (const int*)  d_in[1];
  const int*   dst      = (const int*)  d_in[2];
  const int*   etype    = (const int*)  d_in[3];
  const float* emb      = (const float*)d_in[4];
  const float* basis1   = (const float*)d_in[5];
  const float* comp1    = (const float*)d_in[6];
  const float* loop_w1  = (const float*)d_in[7];
  const float* bias1    = (const float*)d_in[8];
  const float* basis2   = (const float*)d_in[9];
  const float* comp2    = (const float*)d_in[10];
  const float* loop_w2  = (const float*)d_in[11];
  const float* bias2    = (const float*)d_in[12];
  const int nNodes = in_sizes[0];
  const int nEdges = in_sizes[1];

  // workspace layout (in float units, all chunks 16-float = 64B aligned);
  // layer-2 xb/agg overlay layer-1 buffers
  float* ws = (float*)d_ws;
  size_t off = 0;
  __bf16* bt1bf = (__bf16*)(ws + off); off += (size_t)NC1 * H_DIM / 2;
  __bf16* bt2bf = (__bf16*)(ws + off); off += (size_t)NC2 * H_DIM / 2;
  float*  deg   = ws + off; off += (size_t)nNodes; off = (off + 15) & ~(size_t)15;
  __bf16* abf   = (__bf16*)(ws + off); off += (size_t)nNodes * H_DIM / 2;
  __bf16* hbf   = (__bf16*)(ws + off); off += (size_t)nNodes * H_DIM / 2;
  float*  xb    = ws + off; off += (size_t)nNodes * NC1;   // XB1, reused as XB2
  float*  agg   = ws + off; off += (size_t)nNodes * H_DIM; // agg1, reused as agg2
  if (off * sizeof(float) > ws_size) return;               // insufficient scratch

  const int tilesM = (nNodes + 15) / 16;

  // pack / convert operands to bf16
  pack_bt1<<<(NC1 * H_DIM + 255) / 256, 256, 0, stream>>>(basis1, loop_w1, bt1bf);
  pack_bt2<<<(NC2 * H_DIM + 255) / 256, 256, 0, stream>>>(basis2, loop_w2, bt2bf);
  gather_cvt_bf16<<<((size_t)nNodes * H_DIM + 255) / 256, 256, 0, stream>>>(
      emb, node_ids, abf, nNodes);

  // layer 1: XB1 = x @ [basis-cat | loop_w1]
  wmma_gemm_k128<<<dim3(tilesM, NC1 / 16), 32, 0, stream>>>(abf, bt1bf, xb,
                                                            nNodes, NC1);
  hipMemsetAsync(agg, 0, (size_t)nNodes * H_DIM * sizeof(float), stream);
  hipMemsetAsync(deg, 0, (size_t)nNodes * sizeof(float), stream);
  edge_scatter1<<<((size_t)nEdges * 32 + 255) / 256, 256, 0, stream>>>(
      src, dst, etype, xb, comp1, agg, deg, nEdges);
  finish1<<<((size_t)nNodes * H_DIM + 255) / 256, 256, 0, stream>>>(
      agg, deg, xb, bias1, hbf, nNodes);

  // layer 2: XB2 = h @ [basis-cat | loop_w2]
  wmma_gemm_k128<<<dim3(tilesM, NC2 / 16), 32, 0, stream>>>(hbf, bt2bf, xb,
                                                            nNodes, NC2);
  hipMemsetAsync(agg, 0, (size_t)nNodes * OUT_DIM * sizeof(float), stream);
  edge_scatter2<<<((size_t)nEdges * OUT_DIM + 255) / 256, 256, 0, stream>>>(
      src, dst, etype, xb, comp2, agg, nEdges);
  finish2<<<((size_t)nNodes * OUT_DIM + 255) / 256, 256, 0, stream>>>(
      agg, deg, xb, bias2, (float*)d_out, nNodes);
}
